// MosaicBlockLayer_76003741270202
// MI455X (gfx1250) — matrix operand order repaired and checked by reading, not compiled, not run on hardware
//
#include <hip/hip_runtime.h>
#include <cstdint>
#include <cstddef>

// ---------------------------------------------------------------------------
// MosaicBlockLayer for MI455X (gfx1250, wave32, WMMA).
// Heavy GEMMs: v_wmma_f32_16x16x32_bf16, f32 accumulate.
// B tiles staged to LDS with global_load_async_to_lds_b128 (ASYNCcnt),
// double-buffered; A fragments double-buffered in VGPRs; sched_group_barrier
// pins the pipeline to "16 ds_load_b128 then 8 WMMA" per k-step so a single
// s_wait_dscnt covers the whole WMMA burst.
// ---------------------------------------------------------------------------

typedef __attribute__((ext_vector_type(16))) __bf16 bf16x16;
typedef __attribute__((ext_vector_type(8)))  float  v8f;
typedef __attribute__((ext_vector_type(4)))  float  v4f;
typedef __attribute__((ext_vector_type(4)))  int    v4i;

#define AS1 __attribute__((address_space(1)))
#define AS3 __attribute__((address_space(3)))

#if __has_builtin(__builtin_amdgcn_global_load_async_to_lds_b128) && \
    __has_builtin(__builtin_amdgcn_s_wait_asynccnt)
#define USE_ASYNC_LDS 1
#else
#define USE_ASYNC_LDS 0
#endif

#if __has_builtin(__builtin_amdgcn_sched_group_barrier)
#define SCHED_DS_THEN_WMMA(nds, nmm)                       \
  do {                                                     \
    __builtin_amdgcn_sched_group_barrier(0x100, nds, 0);   \
    __builtin_amdgcn_sched_group_barrier(0x008, nmm, 0);   \
  } while (0)
#else
#define SCHED_DS_THEN_WMMA(nds, nmm) do {} while (0)
#endif

static constexpr int Bn  = 4;
static constexpr int Tn  = 2048;
static constexpr int Dn  = 1024;
static constexpr int Kn  = 8;
static constexpr int KCn = 4;
static constexpr int Hn  = 4 * Dn;      // 4096
static constexpr int Mn  = Bn * Tn;     // 8192 activation rows
static constexpr int KDn = Kn * Dn;     // 8192

__device__ __forceinline__ unsigned short f2bf(float f) {
  union { float f; unsigned int u; } c; c.f = f;
  unsigned int u = c.u;
  u += 0x7fffu + ((u >> 16) & 1u);      // round-to-nearest-even
  return (unsigned short)(u >> 16);
}
__device__ __forceinline__ float bf2f(unsigned short h) {
  union { unsigned int u; float f; } c; c.u = ((unsigned int)h) << 16;
  return c.f;
}
__device__ __forceinline__ float sigm(float x) { return 1.0f / (1.0f + __expf(-x)); }

// ---------------------------------------------------------------------------
// Weight prep: W (Krows x N, f32 row-major) -> Wt (N x Krows, bf16 row-major)
// ---------------------------------------------------------------------------
__global__ __launch_bounds__(256)
void wtrans_kernel(const float* __restrict__ W, unsigned short* __restrict__ Wt,
                   int Krows, int N) {
  size_t i = (size_t)blockIdx.x * 256 + threadIdx.x;
  if (i >= (size_t)Krows * N) return;
  int k = (int)(i / N);
  int n = (int)(i % N);
  Wt[(size_t)n * Krows + k] = f2bf(W[i]);
}

// ---------------------------------------------------------------------------
// RMS norm: per row m of (Mn x Dn). Writes f32 and bf16 copies of u.
// ---------------------------------------------------------------------------
__global__ __launch_bounds__(256)
void rmsnorm_kernel(const float* __restrict__ x, float* __restrict__ u,
                    unsigned short* __restrict__ ub) {
  __shared__ float red[256];
  const int m = blockIdx.x;
  const float* xr = x + (size_t)m * Dn;
  float ss = 0.f;
  for (int d = threadIdx.x; d < Dn; d += 256) { float v = xr[d]; ss += v * v; }
  red[threadIdx.x] = ss;
  __syncthreads();
  for (int s = 128; s > 0; s >>= 1) {
    if (threadIdx.x < s) red[threadIdx.x] += red[threadIdx.x + s];
    __syncthreads();
  }
  const float r = rsqrtf(red[0] / (float)Dn + 1e-6f);
  for (int d = threadIdx.x; d < Dn; d += 256) {
    float v = xr[d] * r;
    u[(size_t)m * Dn + d]  = v;
    ub[(size_t)m * Dn + d] = f2bf(v);
  }
}

// ---------------------------------------------------------------------------
// Causal depthwise conv (KC=4): xc[b,t,d] = sum_j w[d,j] * u[b,t-3+j,d]
// ---------------------------------------------------------------------------
__global__ __launch_bounds__(256)
void dwconv_kernel(const float* __restrict__ u, const float* __restrict__ cw,
                   float* __restrict__ xc, unsigned short* __restrict__ xcb) {
  size_t i = (size_t)blockIdx.x * 256 + threadIdx.x;   // over Mn*Dn
  if (i >= (size_t)Mn * Dn) return;
  int d = (int)(i % Dn);
  int m = (int)(i / Dn);
  int t = m % Tn;
  float acc = 0.f;
#pragma unroll
  for (int j = 0; j < KCn; ++j) {
    int tt = t - (KCn - 1) + j;
    if (tt >= 0) acc += cw[d * KCn + j] * u[i - (size_t)(KCn - 1 - j) * Dn];
  }
  xc[i]  = acc;
  xcb[i] = f2bf(acc);
}

// ---------------------------------------------------------------------------
// Tiled bf16 WMMA GEMM: C(M x N) = A(M x K) * Bt(N x K)^T, fused epilogues.
// Block = 256 threads (8 waves). Wave w owns rows [by*128 + 16w, +16),
// block owns columns [bx*128, +128) as eight 16x16 WMMA tiles per wave.
// B tile (128 x 32 bf16, 8 KB) double-buffered in LDS via async-to-LDS;
// all 8 B fragments batch-loaded to VGPRs (sched_group_barrier enforced),
// then 8 back-to-back WMMAs. Fragment layouts per CDNA5 ISA 7.12.2.
// ---------------------------------------------------------------------------
enum { EPI_GATE = 0, EPI_SILU = 1, EPI_BIASF32 = 2, EPI_BF16 = 3, EPI_FINAL = 4 };

template <int EPI>
__global__ __launch_bounds__(256)
void gemm_wmma_bf16(const unsigned short* __restrict__ A,
                    const unsigned short* __restrict__ Bt,
                    int Kd, int Nd,
                    const float* __restrict__ bias,
                    const float* __restrict__ aux0,   // GATE: xc_f32 | FINAL: x
                    const float* __restrict__ aux1,   // FINAL: local
                    const float* __restrict__ aux2,   // FINAL: reads
                    const float* __restrict__ gl,     // FINAL: g_long per row
                    const float* __restrict__ gr,     // FINAL: g_reg per row
                    float* __restrict__ outF,
                    unsigned short* __restrict__ outB) {
  __shared__ unsigned short Bs[2][128][32];           // 16 KB double buffer

  const int tid   = threadIdx.x;
  const int lane  = tid & 31;
  const int wave  = tid >> 5;
  const int mtile = blockIdx.y * 128 + wave * 16;
  const int nbase = blockIdx.x * 128;
  const int lm = lane & 15;   // row within A-tile / column within B-tile
  const int lh = lane >> 4;   // half selector

  v8f acc[8] = {};

  const unsigned short* rowA = A + (size_t)(mtile + lm) * Kd;

  // Stage one 128(n) x 32(k) bf16 B tile into LDS buffer `bufi`.
  // 512 16-byte chunks, 2 per thread.
  auto stageB = [&](int k0, int bufi) {
#pragma unroll
    for (int i = 0; i < 2; ++i) {
      const int c  = tid + i * 256;
      const int n  = c >> 2;
      const int kc = (c & 3) * 8;
      const unsigned short* g = Bt + (size_t)(nbase + n) * Kd + k0 + kc;
#if USE_ASYNC_LDS
      __builtin_amdgcn_global_load_async_to_lds_b128(
          (AS1 v4i*)g, (AS3 v4i*)&Bs[bufi][n][kc], 0, 0);
#else
      *(v4f*)&Bs[bufi][n][kc] = *(const v4f*)g;
#endif
    }
  };

  union ABf { bf16x16 v; v4f q[2]; };
  ABf aCur, aNext;

  // Prologue: stage first B tile, load first A fragment.
  stageB(0, 0);
  aCur.q[0] = *(const v4f*)(rowA + lh * 8);
  aCur.q[1] = *(const v4f*)(rowA + 16 + lh * 8);

  int buf = 0;
  for (int k0 = 0; k0 < Kd; k0 += 32) {
    const int knext = k0 + 32;
    if (knext < Kd) {
      stageB(knext, buf ^ 1);                       // overlaps with compute
      aNext.q[0] = *(const v4f*)(rowA + knext + lh * 8);
      aNext.q[1] = *(const v4f*)(rowA + knext + 16 + lh * 8);
      __builtin_prefetch(rowA + knext + 96, 0, 3);  // near-cache prefetch
    }
#if USE_ASYNC_LDS
    if (knext < Kd) __builtin_amdgcn_s_wait_asynccnt(2);  // next tile in flight
    else            __builtin_amdgcn_s_wait_asynccnt(0);
#endif
    __syncthreads();                                // tile `buf` visible to all

    // Batch-load all 8 B fragments (16 ds_load_b128), one DS wait total,
    // then issue 8 WMMAs back-to-back. sched_group_barrier pins this order.
    ABf bf[8];
#pragma unroll
    for (int j = 0; j < 8; ++j) {
      const unsigned short* lb = &Bs[buf][j * 16 + lm][lh * 8];
      bf[j].q[0] = *(const v4f*)(lb);
      bf[j].q[1] = *(const v4f*)(lb + 16);
    }
#pragma unroll
    for (int j = 0; j < 8; ++j) {
      acc[j] = __builtin_amdgcn_wmma_f32_16x16x32_bf16(
          /*neg_a=*/false, aCur.v, /*neg_b=*/false, bf[j].v,
          /*c_mod=*/(short)0, acc[j], /*reuse_a=*/false, /*reuse_b=*/false);
    }
    SCHED_DS_THEN_WMMA(16, 8);                      // DS-read x16 then WMMA x8
    __syncthreads();                                // done reading before reuse
    aCur = aNext;
    buf ^= 1;
  }

  // C/D layout: VGPR r -> M = r + 8*lh, N = lm (per lane half)
#pragma unroll
  for (int j = 0; j < 8; ++j) {
    const int n = nbase + j * 16 + lm;
#pragma unroll
    for (int r = 0; r < 8; ++r) {
      const int m = mtile + lh * 8 + r;
      const float a = acc[j][r];
      const size_t idx = (size_t)m * Nd + n;
      if constexpr (EPI == EPI_GATE) {
        float pre = a + bias[n];
        outB[idx] = f2bf(aux0[idx] * sigm(pre));          // xg = xc * sigmoid(.)
      } else if constexpr (EPI == EPI_SILU) {
        float pre = a + bias[n];
        outB[idx] = f2bf(pre * sigm(pre));                // silu
      } else if constexpr (EPI == EPI_BIASF32) {
        outF[idx] = a + bias[n];
      } else if constexpr (EPI == EPI_BF16) {
        outB[idx] = f2bf(a);                              // z (no bias)
      } else {                                            // EPI_FINAL
        outF[idx] = aux0[idx] + aux1[idx] + gl[m] * a + gr[m] * aux2[idx];
      }
    }
  }
}

// ---------------------------------------------------------------------------
// Small projections per row m: sel = softmax(u@rs_w + rs_b) (R=16),
// wg / g_long / g_reg = sigmoid of 1-wide projections.
// ---------------------------------------------------------------------------
__global__ __launch_bounds__(128)
void smallproj_kernel(const float* __restrict__ u,
                      const float* __restrict__ rs_w, const float* __restrict__ rs_b,
                      const float* __restrict__ rwg_w, const float* __restrict__ rwg_b,
                      const float* __restrict__ gl_w, const float* __restrict__ gl_b,
                      const float* __restrict__ gr_w, const float* __restrict__ gr_b,
                      float* __restrict__ sel, float* __restrict__ wg,
                      float* __restrict__ gl, float* __restrict__ gr) {
  __shared__ float part[19][128];
  const int m = blockIdx.x;
  const int tid = threadIdx.x;
  float acc[19];
#pragma unroll
  for (int i = 0; i < 19; ++i) acc[i] = 0.f;
  const float* ur = u + (size_t)m * Dn;
  for (int d = tid; d < Dn; d += 128) {
    float ud = ur[d];
#pragma unroll
    for (int r = 0; r < 16; ++r) acc[r] += ud * rs_w[d * 16 + r];
    acc[16] += ud * rwg_w[d];
    acc[17] += ud * gl_w[d];
    acc[18] += ud * gr_w[d];
  }
#pragma unroll
  for (int i = 0; i < 19; ++i) part[i][tid] = acc[i];
  __syncthreads();
  for (int s = 64; s > 0; s >>= 1) {
    if (tid < s)
      for (int i = 0; i < 19; ++i) part[i][tid] += part[i][tid + s];
    __syncthreads();
  }
  if (tid == 0) {
    float l[16], mx = -1e30f;
    for (int r = 0; r < 16; ++r) { l[r] = part[r][0] + rs_b[r]; mx = fmaxf(mx, l[r]); }
    float sum = 0.f;
    for (int r = 0; r < 16; ++r) { l[r] = __expf(l[r] - mx); sum += l[r]; }
    float inv = 1.f / sum;
    for (int r = 0; r < 16; ++r) sel[(size_t)m * 16 + r] = l[r] * inv;
    wg[m] = sigm(part[16][0] + rwg_b[0]);
    gl[m] = sigm(part[17][0] + gl_b[0]);
    gr[m] = sigm(part[18][0] + gr_b[0]);
  }
}

// ---------------------------------------------------------------------------
// Exponential-decay state scan. One thread per (b,k,d) channel (32768 total):
// s[t] = dec_k * s[t-1] + z[t]. z and s_all stored bf16 (feeds so-GEMM).
// ---------------------------------------------------------------------------
__global__ __launch_bounds__(256)
void decay_scan_kernel(const unsigned short* __restrict__ z,
                       const float* __restrict__ decay_logit,
                       unsigned short* __restrict__ sall) {
  int g = blockIdx.x * 256 + threadIdx.x;     // over Bn*KDn
  if (g >= Bn * KDn) return;
  int b = g / KDn;
  int c = g % KDn;
  int k = c / Dn;
  const float dec = sigm(decay_logit[k]);
  size_t base = (size_t)b * Tn * KDn + c;
  float s = 0.f;
  for (int t = 0; t < Tn; ++t) {
    size_t idx = base + (size_t)t * KDn;
    s = dec * s + bf2f(z[idx]);
    sall[idx] = f2bf(s);
  }
}

// ---------------------------------------------------------------------------
// Soft-register scan. regs[r][d] columns are independent across d, so each
// thread owns one d and keeps its 16 register values in VGPRs; sel/wg are
// uniform scalar loads. read happens before the write (matches reference).
// ---------------------------------------------------------------------------
__global__ __launch_bounds__(256)
void reg_scan_kernel(const float* __restrict__ sel, const float* __restrict__ wg,
                     const float* __restrict__ v, float* __restrict__ reads) {
  const int b = blockIdx.x >> 2;                       // Dn/256 == 4 chunks
  const int d = ((blockIdx.x & 3) << 8) + threadIdx.x;
  float regs[16];
#pragma unroll
  for (int r = 0; r < 16; ++r) regs[r] = 0.f;
  for (int t = 0; t < Tn; ++t) {
    const size_t m = (size_t)b * Tn + t;
    const float w = wg[m];
    const float* sm = sel + m * 16;
    const float vv = v[m * Dn + d];
    float rd = 0.f;
#pragma unroll
    for (int r = 0; r < 16; ++r) {
      const float sr = sm[r];
      rd += sr * regs[r];                              // read old value
      regs[r] += (w * sr) * (vv - regs[r]);            // then write
    }
    reads[m * Dn + d] = rd;
  }
}

// ---------------------------------------------------------------------------
// Host-side orchestration.
// ---------------------------------------------------------------------------
extern "C" void kernel_launch(void* const* d_in, const int* in_sizes, int n_in,
                              void* d_out, int out_size, void* d_ws, size_t ws_size,
                              hipStream_t stream) {
  (void)in_sizes; (void)n_in; (void)out_size; (void)ws_size;
  const float* x       = (const float*)d_in[0];
  const float* conv_w  = (const float*)d_in[1];
  const float* gp_w    = (const float*)d_in[2];
  const float* gp_b    = (const float*)d_in[3];
  const float* up_w    = (const float*)d_in[4];
  const float* up_b    = (const float*)d_in[5];
  const float* dn_w    = (const float*)d_in[6];
  const float* dn_b    = (const float*)d_in[7];
  const float* si_w    = (const float*)d_in[8];
  const float* so_w    = (const float*)d_in[9];
  const float* dec_lg  = (const float*)d_in[10];
  const float* rs_w    = (const float*)d_in[11];
  const float* rs_b    = (const float*)d_in[12];
  const float* rwg_w   = (const float*)d_in[13];
  const float* rwg_b   = (const float*)d_in[14];
  const float* rv_w    = (const float*)d_in[15];
  const float* rv_b    = (const float*)d_in[16];
  const float* gr_w    = (const float*)d_in[17];
  const float* gr_b    = (const float*)d_in[18];
  const float* gl_w    = (const float*)d_in[19];
  const float* gl_b    = (const float*)d_in[20];
  float* out = (float*)d_out;

  char* base = (char*)d_ws;
  size_t off = 0;
  auto alloc = [&](size_t bytes) -> char* {
    char* p = base + off;
    off += (bytes + 255) & ~(size_t)255;
    return p;
  };

  const size_t MD  = (size_t)Mn * Dn;
  const size_t MH  = (size_t)Mn * Hn;
  const size_t MKD = (size_t)Mn * KDn;

  float*          u_f     = (float*)alloc(MD * 4);
  unsigned short* u_b     = (unsigned short*)alloc(MD * 2);
  float*          xc_f    = (float*)alloc(MD * 4);
  unsigned short* xc_b    = (unsigned short*)alloc(MD * 2);
  unsigned short* xg_b    = (unsigned short*)alloc(MD * 2);
  unsigned short* h_b     = (unsigned short*)alloc(MH * 2);
  float*          local_f = (float*)alloc(MD * 4);
  unsigned short* z_b     = (unsigned short*)alloc(MKD * 2);
  unsigned short* s_b     = (unsigned short*)alloc(MKD * 2);
  float*          v_f     = (float*)alloc(MD * 4);
  float*          reads_f = (float*)alloc(MD * 4);
  float*          sel_f   = (float*)alloc((size_t)Mn * 16 * 4);
  float*          wg_f    = (float*)alloc((size_t)Mn * 4);
  float*          gl_f    = (float*)alloc((size_t)Mn * 4);
  float*          gr_f    = (float*)alloc((size_t)Mn * 4);
  unsigned short* gp_t    = (unsigned short*)alloc((size_t)Dn * Dn * 2);
  unsigned short* up_t    = (unsigned short*)alloc((size_t)Dn * Hn * 2);
  unsigned short* dn_t    = (unsigned short*)alloc((size_t)Hn * Dn * 2);
  unsigned short* si_t    = (unsigned short*)alloc((size_t)Dn * KDn * 2);
  unsigned short* so_t    = (unsigned short*)alloc((size_t)KDn * Dn * 2);
  unsigned short* rv_t    = (unsigned short*)alloc((size_t)Dn * Dn * 2);

  const dim3 blk256(256);
  auto gridN = [](size_t n) { return dim3((unsigned)((n + 255) / 256)); };

  // --- weight prep: bf16 + transpose to (N x K) ---
  wtrans_kernel<<<gridN((size_t)Dn * Dn),  blk256, 0, stream>>>(gp_w, gp_t, Dn, Dn);
  wtrans_kernel<<<gridN((size_t)Dn * Hn),  blk256, 0, stream>>>(up_w, up_t, Dn, Hn);
  wtrans_kernel<<<gridN((size_t)Hn * Dn),  blk256, 0, stream>>>(dn_w, dn_t, Hn, Dn);
  wtrans_kernel<<<gridN((size_t)Dn * KDn), blk256, 0, stream>>>(si_w, si_t, Dn, KDn);
  wtrans_kernel<<<gridN((size_t)KDn * Dn), blk256, 0, stream>>>(so_w, so_t, KDn, Dn);
  wtrans_kernel<<<gridN((size_t)Dn * Dn),  blk256, 0, stream>>>(rv_w, rv_t, Dn, Dn);

  // --- u = rms_norm(x); xc = causal depthwise conv(u) ---
  rmsnorm_kernel<<<dim3(Mn), blk256, 0, stream>>>(x, u_f, u_b);
  dwconv_kernel<<<gridN(MD), blk256, 0, stream>>>(u_f, conv_w, xc_f, xc_b);

  // --- local mixer GEMM chain (all WMMA bf16) ---
  gemm_wmma_bf16<EPI_GATE><<<dim3(Dn / 128, Mn / 128), blk256, 0, stream>>>(
      xc_b, gp_t, Dn, Dn, gp_b, xc_f, nullptr, nullptr, nullptr, nullptr,
      nullptr, xg_b);
  gemm_wmma_bf16<EPI_SILU><<<dim3(Hn / 128, Mn / 128), blk256, 0, stream>>>(
      xg_b, up_t, Dn, Hn, up_b, nullptr, nullptr, nullptr, nullptr, nullptr,
      nullptr, h_b);
  gemm_wmma_bf16<EPI_BIASF32><<<dim3(Dn / 128, Mn / 128), blk256, 0, stream>>>(
      h_b, dn_t, Hn, Dn, dn_b, nullptr, nullptr, nullptr, nullptr, nullptr,
      local_f, nullptr);

  // --- state path projections ---
  gemm_wmma_bf16<EPI_BF16><<<dim3(KDn / 128, Mn / 128), blk256, 0, stream>>>(
      u_b, si_t, Dn, KDn, nullptr, nullptr, nullptr, nullptr, nullptr, nullptr,
      nullptr, z_b);
  gemm_wmma_bf16<EPI_BIASF32><<<dim3(Dn / 128, Mn / 128), blk256, 0, stream>>>(
      u_b, rv_t, Dn, Dn, rv_b, nullptr, nullptr, nullptr, nullptr, nullptr,
      v_f, nullptr);
  smallproj_kernel<<<dim3(Mn), dim3(128), 0, stream>>>(
      u_f, rs_w, rs_b, rwg_w, rwg_b, gl_w, gl_b, gr_w, gr_b,
      sel_f, wg_f, gl_f, gr_f);

  // --- sequential scans (parallel over channels) ---
  decay_scan_kernel<<<gridN((size_t)Bn * KDn), blk256, 0, stream>>>(z_b, dec_lg, s_b);
  reg_scan_kernel<<<dim3(Bn * (Dn / 256)), blk256, 0, stream>>>(sel_f, wg_f, v_f, reads_f);

  // --- long_out GEMM fused with final combine -> d_out ---
  gemm_wmma_bf16<EPI_FINAL><<<dim3(Dn / 128, Mn / 128), blk256, 0, stream>>>(
      s_b, so_t, KDn, Dn, nullptr, x, local_f, reads_f, gl_f, gr_f,
      out, nullptr);
}
